// CausalSelfMQA_31559419691182
// MI455X (gfx1250) — matrix operand order repaired and checked by reading, not compiled, not run on hardware
//
#include <hip/hip_runtime.h>

#define SEQ    2048
#define BATCH  2
#define DMODEL 2048
#define NHEADS 16
#define DHEAD  128
#define NTOK   (SEQ * BATCH)          // 4096 rows (s*B + b)
#define NQKV   (DMODEL + 2 * DHEAD)   // 2304 fused qkv columns

typedef __attribute__((ext_vector_type(16))) __bf16 v16bf;
typedef __attribute__((ext_vector_type(8)))  float  v8f;

union FragU {
  v16bf bf;
  uint4 q[2];
};

__device__ __forceinline__ unsigned short f2bf(float x) {
  unsigned int u = __float_as_uint(x);
  u += 0x7FFFu + ((u >> 16) & 1u);   // round-to-nearest-even
  return (unsigned short)(u >> 16);
}

__device__ __forceinline__ v8f wmma_bf16(const FragU& a, const FragU& b, v8f c) {
  return __builtin_amdgcn_wmma_f32_16x16x32_bf16(false, a.bf, false, b.bf,
                                                 (short)0, c, false, false);
}

// A-fragment: 16 contiguous bf16 (K = klo + 0..15)
__device__ __forceinline__ void ldA(FragU& f, const unsigned short* p) {
  f.q[0] = *(const uint4*)(p);
  f.q[1] = *(const uint4*)(p + 8);
}
// B-fragment: 16 contiguous bf16 (K = 16*(lane/16) + 0..15)
__device__ __forceinline__ void ldB(FragU& f, const unsigned short* p) {
  f.q[0] = *(const uint4*)(p);
  f.q[1] = *(const uint4*)(p + 8);
}

// ---------------------------------------------------------------- cast f32->bf16
__global__ void cast_f32_bf16(const float* __restrict__ in,
                              unsigned short* __restrict__ out, int n) {
  int i = blockIdx.x * blockDim.x + threadIdx.x;
  int stride = gridDim.x * blockDim.x;
  for (; i < n; i += stride) out[i] = f2bf(in[i]);
}

// ---------------------------------------------------------------- fused QKV GEMM
// C[4096 x 2304] = H_bf16 @ Wqkv^T ; 32x64 tile/wave, double-buffered fragments
__global__ void __launch_bounds__(128) qkv_gemm(
    const unsigned short* __restrict__ hbf,
    const unsigned short* __restrict__ qwbf,
    const unsigned short* __restrict__ kvwbf,
    const float* __restrict__ qb, const float* __restrict__ kvb,
    unsigned short* __restrict__ qbuf,     // (b,h,s,d)
    unsigned short* __restrict__ kbuf,     // (b,s,d)
    unsigned short* __restrict__ vtbuf) {  // (b,d,s)
  const int NT = NQKV / 64;  // 36 n-tiles of 64
  int wg   = blockIdx.x * (blockDim.x >> 5) + (threadIdx.x >> 5);
  int lane = threadIdx.x & 31;
  int mt = wg / NT, nt = wg % NT;
  int m0 = mt * 32, n0 = nt * 64;
  int c = lane & 15, half = lane >> 4, klo = half * 16;

  const unsigned short* arow0 = hbf + (size_t)(m0 + c) * DMODEL + klo;
  const unsigned short* arow1 = arow0 + (size_t)16 * DMODEL;
  const unsigned short* brow[4];
#pragma unroll
  for (int t = 0; t < 4; ++t) {
    int col = n0 + t * 16 + c;
    brow[t] = ((col < DMODEL) ? qwbf + (size_t)col * DMODEL
                              : kvwbf + (size_t)(col - DMODEL) * DMODEL) + half * 16;
  }

  v8f acc[2][4] = {};
  FragU a0[2], a1[2], bb[2][4];
  auto loadStage = [&](int s, int kk) {
    ldA(a0[s], arow0 + kk);
    ldA(a1[s], arow1 + kk);
#pragma unroll
    for (int t = 0; t < 4; ++t) ldB(bb[s][t], brow[t] + kk);
  };
  auto compute = [&](int s) {
#pragma unroll
    for (int t = 0; t < 4; ++t) {
      acc[0][t] = wmma_bf16(a0[s], bb[s][t], acc[0][t]);
      acc[1][t] = wmma_bf16(a1[s], bb[s][t], acc[1][t]);
    }
  };

  loadStage(0, 0);
  for (int kk = 0; kk < DMODEL; kk += 64) {
    loadStage(1, kk + 32);
    compute(0);
    loadStage(0, (kk + 64 < DMODEL) ? kk + 64 : 0);  // last one redundant, harmless
    compute(1);
  }

#pragma unroll
  for (int t = 0; t < 4; ++t) {
    int col = n0 + t * 16 + c;
    float bias = (col < DMODEL) ? qb[col] : kvb[col - DMODEL];
#pragma unroll
    for (int mh = 0; mh < 2; ++mh) {
#pragma unroll
      for (int r = 0; r < 8; ++r) {
        int mrow = m0 + mh * 16 + r + half * 8;  // C layout row
        int s = mrow >> 1, b = mrow & 1;         // flat row = s*B + b
        unsigned short bv = f2bf(acc[mh][t][r] + bias);
        if (col < DMODEL) {
          int h = col >> 7, d = col & 127;
          qbuf[(((size_t)(b * NHEADS + h)) * SEQ + s) * DHEAD + d] = bv;
        } else if (col < DMODEL + DHEAD) {
          kbuf[((size_t)b * SEQ + s) * DHEAD + (col - DMODEL)] = bv;
        } else {
          vtbuf[((size_t)b * DHEAD + (col - DMODEL - DHEAD)) * SEQ + s] = bv;
        }
      }
    }
  }
}

// ---------------------------------------------------------------- flash attention
// one wave per (b, h, 16-query tile); online softmax; bf16 WMMA, f32 accum
__global__ void __launch_bounds__(128, 1) mqa_flash(
    const unsigned short* __restrict__ qbuf,
    const unsigned short* __restrict__ kbuf,
    const unsigned short* __restrict__ vtbuf,
    const int* __restrict__ seqmask,
    unsigned short* __restrict__ obuf) {   // (s*B+b, h*128+d)
  __shared__ __align__(16) unsigned short ldsP[4 * 16 * 32];
  int warp = threadIdx.x >> 5, lane = threadIdx.x & 31;
  int wg = blockIdx.x * 4 + warp;
  int qt = wg & 127;
  int h  = (wg >> 7) & 15;
  int b  = wg >> 11;
  int q0 = qt * 16;
  int c = lane & 15, half = lane >> 4, klo = half * 16;
  unsigned short* myP = ldsP + warp * (16 * 32);
  const float scale = 0.08838834764831845f;  // 1/sqrt(128)

  // persistent Q A-fragments (16 x 128 over 4 K=32 slabs)
  const unsigned short* qrow =
      qbuf + (((size_t)(b * NHEADS + h)) * SEQ + (q0 + c)) * DHEAD + klo;
  FragU aQ[4];
#pragma unroll
  for (int dk = 0; dk < 4; ++dk) ldA(aQ[dk], qrow + dk * 32);

  v8f acc[8] = {};
  float mrow[8], lrow[8];
#pragma unroll
  for (int r = 0; r < 8; ++r) { mrow[r] = -1e30f; lrow[r] = 0.f; }

  const unsigned short* kbase_ptr = kbuf + ((size_t)b * SEQ + c) * DHEAD + half * 16;
  const int kend = q0 + 16;
  for (int kbase = 0; kbase < kend; kbase += 32) {
    // ---- Q @ K^T : preload all 8 B-fragments (one clause), then 8 WMMAs
    const unsigned short* kp0 = kbase_ptr + (size_t)kbase * DHEAD;
    const unsigned short* kp1 = kp0 + (size_t)16 * DHEAD;
    FragU kb0[4], kb1[4];
#pragma unroll
    for (int dk = 0; dk < 4; ++dk) {
      ldB(kb0[dk], kp0 + dk * 32);
      ldB(kb1[dk], kp1 + dk * 32);
    }
    // prefetch next chunk's K rows while this chunk computes
    {
      int nk = (kbase + 32 < kend) ? kbase + 32 : kbase;
      const unsigned short* np = kbase_ptr + (size_t)nk * DHEAD;
      __builtin_prefetch(np, 0, 1);
      __builtin_prefetch(np + 16 * DHEAD, 0, 1);
    }
    v8f s0 = {}, s1 = {};
#pragma unroll
    for (int dk = 0; dk < 4; ++dk) {
      s0 = wmma_bf16(aQ[dk], kb0[dk], s0);
      s1 = wmma_bf16(aQ[dk], kb1[dk], s1);
    }

    // ---- causal + sequence mask, row reductions, online softmax
    int key0 = kbase + c, key1 = kbase + 16 + c;
    int sm0 = seqmask[b * SEQ + key0];
    int sm1 = seqmask[b * SEQ + key1];
    float alpha[8];
#pragma unroll
    for (int r = 0; r < 8; ++r) {
      int qg = q0 + r + half * 8;
      float v0 = (key0 <= qg && sm0) ? s0[r] * scale : -1e30f;
      float v1 = (key1 <= qg && sm1) ? s1[r] * scale : -1e30f;
      float mx = fmaxf(v0, v1);
      mx = fmaxf(mx, __shfl_xor(mx, 1));
      mx = fmaxf(mx, __shfl_xor(mx, 2));
      mx = fmaxf(mx, __shfl_xor(mx, 4));
      mx = fmaxf(mx, __shfl_xor(mx, 8));
      float nm = fmaxf(mrow[r], mx);
      float p0 = __expf(v0 - nm);
      float p1 = __expf(v1 - nm);
      s0[r] = p0; s1[r] = p1;
      float rs = p0 + p1;
      rs += __shfl_xor(rs, 1);
      rs += __shfl_xor(rs, 2);
      rs += __shfl_xor(rs, 4);
      rs += __shfl_xor(rs, 8);
      float a = __expf(mrow[r] - nm);
      alpha[r] = a;
      lrow[r] = lrow[r] * a + rs;
      mrow[r] = nm;
    }
#pragma unroll
    for (int db = 0; db < 8; ++db)
#pragma unroll
      for (int r = 0; r < 8; ++r) acc[db][r] *= alpha[r];

    // ---- P: C-fragment (f32) -> LDS (bf16) -> A-fragment re-layout (same wave)
#pragma unroll
    for (int r = 0; r < 8; ++r) {
      int prow = r + half * 8;
      myP[prow * 32 + c]      = f2bf(s0[r]);
      myP[prow * 32 + 16 + c] = f2bf(s1[r]);
    }
    FragU pa;
    ldA(pa, myP + c * 32 + klo);

    // ---- P(16x32) @ V(32x128): preload 8 B-fragments, then 8 WMMAs
    FragU vb[8];
#pragma unroll
    for (int db = 0; db < 8; ++db)
      ldB(vb[db], vtbuf + ((size_t)b * DHEAD + db * 16 + c) * SEQ + kbase + half * 16);
#pragma unroll
    for (int db = 0; db < 8; ++db) acc[db] = wmma_bf16(pa, vb[db], acc[db]);
  }

  // normalize and store (bf16) to (s*B+b, h*128+d)
#pragma unroll
  for (int r = 0; r < 8; ++r) {
    float inv = 1.0f / lrow[r];
    int qg = q0 + r + half * 8;
    size_t base = ((size_t)qg * BATCH + b) * DMODEL + h * DHEAD;
#pragma unroll
    for (int db = 0; db < 8; ++db)
      obuf[base + db * 16 + c] = f2bf(acc[db][r] * inv);
  }
}

// ---------------------------------------------------------------- output GEMM
// out[4096 x 2048] = Obf @ Ow^T + b ; 32x64 tile/wave, double-buffered
__global__ void __launch_bounds__(128) out_gemm(
    const unsigned short* __restrict__ obuf,
    const unsigned short* __restrict__ owbf,
    const float* __restrict__ ob, float* __restrict__ out) {
  const int NT = DMODEL / 64;  // 32
  int wg   = blockIdx.x * (blockDim.x >> 5) + (threadIdx.x >> 5);
  int lane = threadIdx.x & 31;
  int mt = wg / NT, nt = wg % NT;
  int m0 = mt * 32, n0 = nt * 64;
  int c = lane & 15, half = lane >> 4, klo = half * 16;

  const unsigned short* arow0 = obuf + (size_t)(m0 + c) * DMODEL + klo;
  const unsigned short* arow1 = arow0 + (size_t)16 * DMODEL;
  const unsigned short* brow[4];
#pragma unroll
  for (int t = 0; t < 4; ++t)
    brow[t] = owbf + (size_t)(n0 + t * 16 + c) * DMODEL + half * 16;

  v8f acc[2][4] = {};
  FragU a0[2], a1[2], bb[2][4];
  auto loadStage = [&](int s, int kk) {
    ldA(a0[s], arow0 + kk);
    ldA(a1[s], arow1 + kk);
#pragma unroll
    for (int t = 0; t < 4; ++t) ldB(bb[s][t], brow[t] + kk);
  };
  auto compute = [&](int s) {
#pragma unroll
    for (int t = 0; t < 4; ++t) {
      acc[0][t] = wmma_bf16(a0[s], bb[s][t], acc[0][t]);
      acc[1][t] = wmma_bf16(a1[s], bb[s][t], acc[1][t]);
    }
  };

  loadStage(0, 0);
  for (int kk = 0; kk < DMODEL; kk += 64) {
    loadStage(1, kk + 32);
    compute(0);
    loadStage(0, (kk + 64 < DMODEL) ? kk + 64 : 0);
    compute(1);
  }

#pragma unroll
  for (int t = 0; t < 4; ++t) {
    int col = n0 + t * 16 + c;
    float bias = ob[col];
#pragma unroll
    for (int mh = 0; mh < 2; ++mh) {
#pragma unroll
      for (int r = 0; r < 8; ++r) {
        int m = m0 + mh * 16 + r + half * 8;
        out[(size_t)m * DMODEL + col] = acc[mh][t][r] + bias;
      }
    }
  }
}

// ---------------------------------------------------------------- launch
extern "C" void kernel_launch(void* const* d_in, const int* in_sizes, int n_in,
                              void* d_out, int out_size, void* d_ws, size_t ws_size,
                              hipStream_t stream) {
  const float* hidden = (const float*)d_in[0];
  const int*   smask  = (const int*)d_in[1];
  const float* qw  = (const float*)d_in[2];
  const float* qb  = (const float*)d_in[3];
  const float* kvw = (const float*)d_in[4];
  const float* kvb = (const float*)d_in[5];
  const float* ow  = (const float*)d_in[6];
  const float* ob  = (const float*)d_in[7];
  float* out = (float*)d_out;

  char* ws = (char*)d_ws;
  unsigned short* hbf   = (unsigned short*)(ws);              // 16 MB
  unsigned short* qwbf  = (unsigned short*)(ws + 16777216);   //  8 MB
  unsigned short* kvwbf = (unsigned short*)(ws + 25165824);   //  1 MB
  unsigned short* owbf  = (unsigned short*)(ws + 26214400);   //  8 MB
  unsigned short* qbuf  = (unsigned short*)(ws + 34603008);   // 16 MB
  unsigned short* kbuf  = (unsigned short*)(ws + 51380224);   //  1 MB
  unsigned short* vtbuf = (unsigned short*)(ws + 52428800);   //  1 MB
  unsigned short* obuf  = (unsigned short*)(ws + 53477376);   // 16 MB

  cast_f32_bf16<<<256, 256, 0, stream>>>(hidden, hbf, NTOK * DMODEL);
  cast_f32_bf16<<<256, 256, 0, stream>>>(qw, qwbf, DMODEL * DMODEL);
  cast_f32_bf16<<<64, 256, 0, stream>>>(kvw, kvwbf, 2 * DHEAD * DMODEL);
  cast_f32_bf16<<<256, 256, 0, stream>>>(ow, owbf, DMODEL * DMODEL);

  {
    int waves = (NTOK / 32) * (NQKV / 64);   // 4608
    qkv_gemm<<<waves / 4, 128, 0, stream>>>(hbf, qwbf, kvwbf, qb, kvb,
                                            qbuf, kbuf, vtbuf);
  }
  {
    int waves = BATCH * NHEADS * (SEQ / 16); // 4096
    mqa_flash<<<waves / 4, 128, 0, stream>>>(qbuf, kbuf, vtbuf, smask, obuf);
  }
  {
    int waves = (NTOK / 32) * (DMODEL / 64); // 4096
    out_gemm<<<waves / 4, 128, 0, stream>>>(obuf, owbf, ob, out);
  }
}
